// EfficientAttention1D_85375359910066
// MI455X (gfx1250) — matrix operand order repaired and checked by reading, not compiled, not run on hardware
//
#include <hip/hip_runtime.h>
#include <hip/hip_bf16.h>
#include <math.h>

// ---------------------------------------------------------------------------
// EfficientAttention1D for MI455X (gfx1250, wave32, WMMA f32_16x16x32_f16)
//
// B=8, C=256, L=8192, H=8, D=64, INNER=512.  ~77 GFLOP, 128MB min HBM traffic
// -> compute-leaning; all matmuls via v_wmma_f32_16x16x32_f16 (f32 accum).
// k-softmax factored as diag(1/Z) * exp(K-max)^T * V: exp fuses into the
// context GEMM's B-fragment, 1/Z folds into the 64x64 context read in K4.
// All intermediates laid out so WMMA fragments load contiguous-K.
// K3 uses the Tensor Data Mover (tensor_load_to_lds, inline asm + packed D#)
// with TDM LDS-padding (pitch 264 halves) and s_wait_tensorcnt pipelining.
// K1/K5 register-double-buffer their fragment loads over the WMMA batches.
// ---------------------------------------------------------------------------

#define DEV __device__ __forceinline__

constexpr int Bn    = 8;
constexpr int Cc    = 256;
constexpr int Ll    = 8192;
constexpr int Hh    = 8;
constexpr int Dd    = 64;
constexpr int INNER = 512;

typedef __attribute__((ext_vector_type(16))) _Float16 v16h;
typedef __attribute__((ext_vector_type(8)))  _Float16 v8h;
typedef __attribute__((ext_vector_type(8)))  float    v8f;
typedef __attribute__((ext_vector_type(4)))  unsigned int su32x4;
typedef __attribute__((ext_vector_type(8)))  unsigned int su32x8;

// D = A(16x32 f16) * B(32x16 f16) + C(16x16 f32)
DEV v8f wmma16(v16h a, v16h b, v8f c) {
  return __builtin_amdgcn_wmma_f32_16x16x32_f16(
      false, a, false, b, (short)0, c, false, false);
}

// A-matrix 16x32 f16: lane<16 -> row=lane, k=0..15
//                     lane>=16 -> row=lane-16, k=16..31
DEV v16h frag_a_f16(const _Float16* row, int lane, int k0) {
  int kb = k0 + ((lane >> 4) << 4);
  v8h lo = *(const v8h*)(row + kb);
  v8h hi = *(const v8h*)(row + kb + 8);
  v16h f;
#pragma unroll
  for (int e = 0; e < 8; ++e) { f[e] = lo[e]; f[e + 8] = hi[e]; }
  return f;
}

DEV v16h frag_a_f32cvt(const float* row, int lane, int k0) {
  int kb = k0 + ((lane >> 4) << 4);
  v16h f;
#pragma unroll
  for (int e = 0; e < 8; ++e) {
    f[e]     = (_Float16)row[kb + e];
    f[e + 8] = (_Float16)row[kb + 8 + e];
  }
  return f;
}

// B-matrix 32x16 f16: lane<16 -> col=lane, k=0..15; lane>=16 -> k=16..31
DEV v16h frag_b_f16(const _Float16* col, int lane, int k0) {
  int kb = k0 + ((lane >> 4) << 4);
  v8h a = *(const v8h*)(col + kb);
  v8h b = *(const v8h*)(col + kb + 8);
  v16h f;
#pragma unroll
  for (int e = 0; e < 8; ++e) { f[e] = a[e]; f[e + 8] = b[e]; }
  return f;
}

// ---------------------------------------------------------------------------
// Tensor Data Mover: 2D tile load (64 rows x 256 f16, row stride L) into LDS
// with TDM padding: +4 DWORDs every 128 DWORDs -> LDS pitch 264 halves.
// D# packed per CDNA5 ISA (group0 128b, group1 256b).
// Inline asm (portable across toolchains, bypasses builtin arity skew).
// ---------------------------------------------------------------------------
DEV void tdm_load_tile64x256(const _Float16* gptr, unsigned lds_byte_off) {
  unsigned long long ga = (unsigned long long)(uintptr_t)gptr;
  su32x4 g0;
  g0[0] = 1u;                                        // count=1, user mode
  g0[1] = lds_byte_off;                              // lds_addr
  g0[2] = (unsigned)(ga & 0xFFFFFFFFu);              // global_addr[31:0]
  g0[3] = ((unsigned)(ga >> 32) & 0x01FFFFFFu)       // global_addr[56:32]
          | (2u << 30);                              // type=2 (image)
  su32x8 g1;
  g1[0] = (1u << 16)      // data_size=1 -> 2 bytes
        | (1u << 20)      // pad_enable
        | (6u << 22)      // pad_interval: every 128 DWORDs (= one 256-half row)
        | (3u << 25);     // pad_amount: 4 DWORDs (= 8 halves)
  g1[1] = (unsigned)(Ll & 0xFFFF) << 16;             // tensor_dim0 lo16 (8192)
  g1[2] = (unsigned)(Ll >> 16)                       // tensor_dim0 hi16
        | ((unsigned)(INNER & 0xFFFF) << 16);        // tensor_dim1 lo16 (512)
  g1[3] = 0u | (256u << 16);                         // dim1 hi16 | tile_dim0=256
  g1[4] = 64u;                                       // tile_dim1=64, tile_dim2=0
  g1[5] = (unsigned)Ll;                              // tensor_dim0_stride lo32
  g1[6] = 0u;                                        // stride hi | dim1_stride lo
  g1[7] = 0u;
  asm volatile("tensor_load_to_lds %0, %1" :: "s"(g0), "s"(g1) : "memory");
}

// ---------------------------------------------------------------------------
// K0: transpose x (b,c,l) f32 -> xT (b,l,c) f16.
// ---------------------------------------------------------------------------
__global__ void __launch_bounds__(256)
ea_k0_transpose(const float* __restrict__ x, _Float16* __restrict__ xT) {
  __shared__ float tile[64][65];
  int b = blockIdx.x, c0 = blockIdx.y * 64, l0 = blockIdx.z * 64;
  int t = threadIdx.x;
  const float* xb = x + ((size_t)b * Cc + c0) * Ll + l0;
#pragma unroll
  for (int i = 0; i < 16; ++i) {
    int idx = t + i * 256; int r = idx >> 6, cl = idx & 63;
    tile[r][cl] = xb[(size_t)r * Ll + cl];
  }
  __syncthreads();
  _Float16* o = xT + ((size_t)b * Ll + l0) * Cc + c0;
#pragma unroll
  for (int i = 0; i < 16; ++i) {
    int idx = t + i * 256; int r = idx >> 6, cc = idx & 63;
    o[(size_t)r * Cc + cc] = (_Float16)tile[cc][r];
  }
}

// ---------------------------------------------------------------------------
// K1: qkv projection GEMM, register-double-buffered; q softmax fused epilogue.
// ---------------------------------------------------------------------------
__global__ void __launch_bounds__(256)
ea_k1_qkv(const float* __restrict__ Wq, const float* __restrict__ Wk,
          const float* __restrict__ Wv, const _Float16* __restrict__ xT,
          _Float16* __restrict__ qn, _Float16* __restrict__ kr,
          _Float16* __restrict__ vr) {
  int b = blockIdx.x;
  int i0 = blockIdx.y * 64;
  int l0 = blockIdx.z * 128;
  int t = threadIdx.x, w = t >> 5, lane = t & 31;
  int mw = w & 3, nw = w >> 2;

  const float* W; int which, rbase;
  if (i0 < 512)        { W = Wq; which = 0; rbase = i0; }
  else if (i0 < 1024)  { W = Wk; which = 1; rbase = i0 - 512; }
  else                 { W = Wv; which = 2; rbase = i0 - 1024; }

  const float* arow = W + (size_t)(rbase + mw * 16 + (lane & 15)) * Cc;
  const _Float16* brow[4];
#pragma unroll
  for (int nt = 0; nt < 4; ++nt)
    brow[nt] = xT + ((size_t)b * Ll + l0 + nw * 64 + nt * 16 + (lane & 15)) * Cc;

  v8f acc[4];
#pragma unroll
  for (int nt = 0; nt < 4; ++nt) { v8f z = {}; acc[nt] = z; }

  // software pipeline: fragments for step k+1 load before step k's WMMAs
  v16h a_cur = frag_a_f32cvt(arow, lane, 0);
  v16h b_cur[4];
#pragma unroll
  for (int nt = 0; nt < 4; ++nt) b_cur[nt] = frag_b_f16(brow[nt], lane, 0);

#pragma unroll
  for (int k0 = 0; k0 < Cc; k0 += 32) {
    v16h a_nxt = a_cur, b_nxt[4];
#pragma unroll
    for (int nt = 0; nt < 4; ++nt) b_nxt[nt] = b_cur[nt];
    if (k0 + 32 < Cc) {
      int kn = k0 + 32;
      __builtin_prefetch(brow[0] + kn + 32, 0, 3);   // global_prefetch_b8 (WGP)
      a_nxt = frag_a_f32cvt(arow, lane, kn);
#pragma unroll
      for (int nt = 0; nt < 4; ++nt) b_nxt[nt] = frag_b_f16(brow[nt], lane, kn);
    }
#pragma unroll
    for (int nt = 0; nt < 4; ++nt) acc[nt] = wmma16(a_cur, b_cur[nt], acc[nt]);
    a_cur = a_nxt;
#pragma unroll
    for (int nt = 0; nt < 4; ++nt) b_cur[nt] = b_nxt[nt];
  }

  __shared__ float smem[64 * 132];
#pragma unroll
  for (int nt = 0; nt < 4; ++nt) {
    int n = nw * 64 + nt * 16 + (lane & 15);
#pragma unroll
    for (int e = 0; e < 8; ++e) {
      int me = ((lane >> 4) << 3) + e;
      smem[(mw * 16 + me) * 132 + n] = acc[nt][e];
    }
  }
  __syncthreads();

  if (which == 0) {
    int h = i0 >> 6;
    if (t < 128) {
      int col = t;
      float mx = -3.0e38f;
      for (int d = 0; d < 64; ++d) mx = fmaxf(mx, smem[d * 132 + col]);
      float s = 0.f;
      for (int d = 0; d < 64; ++d) s += __expf(smem[d * 132 + col] - mx);
      float inv = 1.f / s;
      _Float16* dst = qn + ((size_t)(b * Hh + h) * Ll + l0 + col) * Dd;
      for (int d = 0; d < 64; ++d)
        dst[d] = (_Float16)(__expf(smem[d * 132 + col] - mx) * inv);
    }
  } else {
    _Float16* dst = (which == 1 ? kr : vr) + ((size_t)(b * INNER + rbase)) * Ll + l0;
    for (int e = t; e < 64 * 128; e += 256) {
      int r = e >> 7, cl = e & 127;
      dst[(size_t)r * Ll + cl] = (_Float16)smem[r * 132 + cl];
    }
  }
}

// ---------------------------------------------------------------------------
// K2: per-row (b, channel) max and 1/sum(exp) over L for k-softmax.
// ---------------------------------------------------------------------------
__global__ void __launch_bounds__(256)
ea_k2_kstats(const _Float16* __restrict__ kr, float* __restrict__ kmax,
             float* __restrict__ kinvz) {
  __shared__ float red[256];
  int row = blockIdx.x, t = threadIdx.x;
  const _Float16* p = kr + (size_t)row * Ll;
  float mx = -3.0e38f;
  for (int i = t; i < Ll; i += 256) mx = fmaxf(mx, (float)p[i]);
  red[t] = mx; __syncthreads();
  for (int s = 128; s > 0; s >>= 1) {
    if (t < s) red[t] = fmaxf(red[t], red[t + s]);
    __syncthreads();
  }
  mx = red[0]; __syncthreads();
  float sum = 0.f;
  for (int i = t; i < Ll; i += 256) sum += __expf((float)p[i] - mx);
  red[t] = sum; __syncthreads();
  for (int s = 128; s > 0; s >>= 1) {
    if (t < s) red[t] += red[t + s];
    __syncthreads();
  }
  if (t == 0) { kmax[row] = mx; kinvz[row] = 1.f / red[0]; }
}

// ---------------------------------------------------------------------------
// K3: ctxT[b,h,m,d] += sum_l v[m,l] * exp(k[d,l]-max_d).  TDM-fed:
//     wave 0 DMAs 64x256 f16 k/v tiles into LDS (double-buffered, TDM pad
//     gives pitch 264), s_wait_tensorcnt pipelines chunk c vs c+1, all 8
//     waves consume via ds_load fragments (wave w owns columns w*32..+31).
//     LDS reduce (csum aliases the dead tile pool) then global f32 atomics.
// ---------------------------------------------------------------------------
constexpr int K3_PITCH  = 264;                   // halves per padded LDS row
constexpr int K3_TILE_H = 64 * K3_PITCH;         // halves per 64x256 tile
constexpr int K3_TILE_B = K3_TILE_H * 2;         // 33792 bytes
constexpr int K3_LDS_B  = 4 * K3_TILE_B + 256;   // k0,v0,k1,v1 + smax

__global__ void __launch_bounds__(256)
ea_k3_context(const _Float16* __restrict__ kr, const _Float16* __restrict__ vr,
              const float* __restrict__ kmax, float* __restrict__ ctxT) {
  __shared__ __align__(16) unsigned char lds_raw[K3_LDS_B];  // pool at LDS off 0
  _Float16* pool = (_Float16*)lds_raw;
  float* smax = (float*)(lds_raw + 4 * K3_TILE_B);

  int bh = blockIdx.x, b = bh >> 3, h = bh & 7;
  int t = threadIdx.x, w = t >> 5, lane = t & 31;
  if (t < 64) smax[t] = kmax[b * INNER + h * 64 + t];

  const _Float16* kbase = kr + ((size_t)(b * INNER + h * 64)) * Ll;
  const _Float16* vbase = vr + ((size_t)(b * INNER + h * 64)) * Ll;
  int lbase = blockIdx.y * 1024;                  // this WG's K-range start

  // prologue: chunk 0 into buffer 0  (wave-uniform branch: TDM ignores EXEC)
  if (w == 0) {
    tdm_load_tile64x256(kbase + lbase, 0u);
    tdm_load_tile64x256(vbase + lbase, (unsigned)K3_TILE_B);
  }

  v8f acc[4][4];
#pragma unroll
  for (int mt = 0; mt < 4; ++mt)
#pragma unroll
    for (int nt = 0; nt < 4; ++nt) { v8f z = {}; acc[mt][nt] = z; }

  for (int c = 0; c < 4; ++c) {
    int cur = c & 1;
    if (w == 0) {
      if (c + 1 < 4) {   // issue next chunk into the other buffer
        unsigned boff = (unsigned)(((c + 1) & 1) * 2 * K3_TILE_B);
        tdm_load_tile64x256(kbase + lbase + (c + 1) * 256, boff);
        tdm_load_tile64x256(vbase + lbase + (c + 1) * 256, boff + K3_TILE_B);
        // chunk c's 2 loads done once <=2 remain in flight (in-order TDM)
        __builtin_amdgcn_s_wait_tensorcnt((short)2);
      } else {
        __builtin_amdgcn_s_wait_tensorcnt((short)0);
      }
    }
    __syncthreads();

    const _Float16* ktile = pool + cur * 2 * K3_TILE_H;
    const _Float16* vtile = ktile + K3_TILE_H;
    int l0w = w * 32;                            // this wave's K-step columns
    v16h a[4], bb[4];
#pragma unroll
    for (int mt = 0; mt < 4; ++mt) {
      const _Float16* row = vtile + (size_t)(mt * 16 + (lane & 15)) * K3_PITCH;
      a[mt] = frag_a_f16(row, lane, l0w);
    }
#pragma unroll
    for (int nt = 0; nt < 4; ++nt) {
      int d = nt * 16 + (lane & 15);
      const _Float16* row = ktile + (size_t)d * K3_PITCH;
      float mxd = smax[d];
      int kb = l0w + ((lane >> 4) << 4);
      v16h f;
#pragma unroll
      for (int e = 0; e < 16; ++e)
        f[e] = (_Float16)__expf((float)row[kb + e] - mxd);
      bb[nt] = f;
    }
#pragma unroll
    for (int mt = 0; mt < 4; ++mt)
#pragma unroll
      for (int nt = 0; nt < 4; ++nt)
        acc[mt][nt] = wmma16(a[mt], bb[nt], acc[mt][nt]);
    __syncthreads();                             // done reading buf[cur]
  }

  // cross-wave reduce: csum aliases the (now dead) tile pool
  float* csum = (float*)lds_raw;
  for (int e = t; e < 4096; e += 256) csum[e] = 0.f;
  __syncthreads();
#pragma unroll
  for (int mt = 0; mt < 4; ++mt)
#pragma unroll
    for (int nt = 0; nt < 4; ++nt)
#pragma unroll
      for (int e = 0; e < 8; ++e) {
        int me = ((lane >> 4) << 3) + e;
        atomicAdd(&csum[(mt * 16 + me) * 64 + nt * 16 + (lane & 15)],
                  acc[mt][nt][e]);
      }
  __syncthreads();
  float* cdst = ctxT + (size_t)bh * 4096;
  for (int e = t; e < 4096; e += 256) atomicAdd(&cdst[e], csum[e]);
}

// ---------------------------------------------------------------------------
// K4: out[l,m] = sum_d qn[l,d] * (invZ_d * ctxT[m,d]); stored (b,l,i) f16.
// ---------------------------------------------------------------------------
__global__ void __launch_bounds__(256)
ea_k4_qctx(const _Float16* __restrict__ qn, const float* __restrict__ ctxT,
           const float* __restrict__ kinvz, _Float16* __restrict__ oi) {
  int bh = blockIdx.x, b = bh >> 3, h = bh & 7;
  int t = threadIdx.x, w = t >> 5, lane = t & 31;
  __shared__ float sinvz[64];
  if (t < 64) sinvz[t] = kinvz[b * INNER + h * 64 + t];
  __syncthreads();

  int l0w = blockIdx.y * 128 + w * 16;
  const float* cbase = ctxT + (size_t)bh * 4096;
  const _Float16* arow = qn + ((size_t)bh * Ll + l0w + (lane & 15)) * Dd;

  v8f acc[4];
#pragma unroll
  for (int nt = 0; nt < 4; ++nt) { v8f z = {}; acc[nt] = z; }

#pragma unroll
  for (int d0 = 0; d0 < 64; d0 += 32) {
    v16h a = frag_a_f16(arow, lane, d0);
#pragma unroll
    for (int nt = 0; nt < 4; ++nt) {
      int m = nt * 16 + (lane & 15);
      const float* crow = cbase + (size_t)m * 64;
      int kb = d0 + ((lane >> 4) << 4);
      v16h f;
#pragma unroll
      for (int e = 0; e < 16; ++e)
        f[e] = (_Float16)(crow[kb + e] * sinvz[kb + e]);
      acc[nt] = wmma16(a, f, acc[nt]);
    }
  }

#pragma unroll
  for (int nt = 0; nt < 4; ++nt) {
    int m = nt * 16 + (lane & 15);
#pragma unroll
    for (int e = 0; e < 8; ++e) {
      int me = ((lane >> 4) << 3) + e;
      int l = l0w + me;
      oi[((size_t)b * Ll + l) * INNER + h * 64 + m] = (_Float16)acc[nt][e];
    }
  }
}

// ---------------------------------------------------------------------------
// K5: y = Wo @ out_inner + bo, register-double-buffered fragment pipeline.
// ---------------------------------------------------------------------------
__global__ void __launch_bounds__(256)
ea_k5_final(const float* __restrict__ Wo, const float* __restrict__ bo,
            const _Float16* __restrict__ oi, float* __restrict__ y) {
  int t = threadIdx.x, w = t >> 5, lane = t & 31;
  int T = blockIdx.x * 8 + w;
  int b = T >> 11;
  int rem = T & 2047;
  int mt = rem >> 7;
  int ntb = rem & 127;
  int c0 = mt * 16, l0 = ntb * 64;

  const float* arow = Wo + (size_t)(c0 + (lane & 15)) * INNER;
  const _Float16* obase = oi + (size_t)b * Ll * INNER;
  const _Float16* brow[4];
#pragma unroll
  for (int nt = 0; nt < 4; ++nt)
    brow[nt] = obase + (size_t)(l0 + nt * 16 + (lane & 15)) * INNER;

  v8f acc[4];
#pragma unroll
  for (int nt = 0; nt < 4; ++nt) { v8f z = {}; acc[nt] = z; }

  v16h a_cur = frag_a_f32cvt(arow, lane, 0);
  v16h b_cur[4];
#pragma unroll
  for (int nt = 0; nt < 4; ++nt) b_cur[nt] = frag_b_f16(brow[nt], lane, 0);

#pragma unroll 4
  for (int k0 = 0; k0 < INNER; k0 += 32) {
    v16h a_nxt = a_cur, b_nxt[4];
#pragma unroll
    for (int nt = 0; nt < 4; ++nt) b_nxt[nt] = b_cur[nt];
    if (k0 + 32 < INNER) {
      int kn = k0 + 32;
      __builtin_prefetch(brow[0] + kn + 32, 0, 3);
      a_nxt = frag_a_f32cvt(arow, lane, kn);
#pragma unroll
      for (int nt = 0; nt < 4; ++nt) b_nxt[nt] = frag_b_f16(brow[nt], lane, kn);
    }
#pragma unroll
    for (int nt = 0; nt < 4; ++nt) acc[nt] = wmma16(a_cur, b_cur[nt], acc[nt]);
    a_cur = a_nxt;
#pragma unroll
    for (int nt = 0; nt < 4; ++nt) b_cur[nt] = b_nxt[nt];
  }

#pragma unroll
  for (int nt = 0; nt < 4; ++nt) {
    int n = l0 + nt * 16 + (lane & 15);
#pragma unroll
    for (int e = 0; e < 8; ++e) {
      int me = ((lane >> 4) << 3) + e;
      int c = c0 + me;
      y[((size_t)b * Cc + c) * Ll + n] = acc[nt][e] + bo[c];
    }
  }
}

// ---------------------------------------------------------------------------
extern "C" void kernel_launch(void* const* d_in, const int* in_sizes, int n_in,
                              void* d_out, int out_size, void* d_ws, size_t ws_size,
                              hipStream_t stream) {
  (void)in_sizes; (void)n_in; (void)out_size; (void)ws_size;
  const float* x  = (const float*)d_in[0];
  const float* Wq = (const float*)d_in[1];
  const float* Wk = (const float*)d_in[2];
  const float* Wv = (const float*)d_in[3];
  const float* Wo = (const float*)d_in[4];
  const float* bo = (const float*)d_in[5];
  float* y = (float*)d_out;

  char* ws = (char*)d_ws;
  size_t off = 0;
  auto alloc = [&](size_t bytes) -> void* {
    void* p = ws + off; off += (bytes + 255) & ~(size_t)255; return p;
  };
  _Float16* xT   = (_Float16*)alloc((size_t)Bn * Ll * Cc * 2);       // 32MB
  _Float16* qn   = (_Float16*)alloc((size_t)Bn * Hh * Ll * Dd * 2);  // 64MB
  _Float16* kr   = (_Float16*)alloc((size_t)Bn * INNER * Ll * 2);    // 64MB
  _Float16* vr   = (_Float16*)alloc((size_t)Bn * INNER * Ll * 2);    // 64MB
  float*    kmax = (float*)alloc((size_t)Bn * INNER * 4);
  float*    kinv = (float*)alloc((size_t)Bn * INNER * 4);
  float*    ctxT = (float*)alloc((size_t)Bn * Hh * Dd * Dd * 4);     // 1MB
  _Float16* oi   = (_Float16*)alloc((size_t)Bn * Ll * INNER * 2);    // 64MB

  ea_k0_transpose<<<dim3(Bn, Cc / 64, Ll / 64), 256, 0, stream>>>(x, xT);
  ea_k1_qkv<<<dim3(Bn, 24, Ll / 128), 256, 0, stream>>>(Wq, Wk, Wv, xT, qn, kr, vr);
  ea_k2_kstats<<<dim3(Bn * INNER), 256, 0, stream>>>(kr, kmax, kinv);
  (void)hipMemsetAsync(ctxT, 0, (size_t)Bn * Hh * Dd * Dd * 4, stream);
  ea_k3_context<<<dim3(Bn * Hh, 8), 256, 0, stream>>>(kr, vr, kmax, ctxT);
  ea_k4_qctx<<<dim3(Bn * Hh, Ll / 128), 256, 0, stream>>>(qn, ctxT, kinv, oi);
  ea_k5_final<<<dim3(2048), 256, 0, stream>>>(Wo, bo, oi, y);
}
